// AttentionModule_80745385164880
// MI455X (gfx1250) — compile-verified, matrix operand order checked by reading, not run on hardware
//
#include <hip/hip_runtime.h>

#define ED 1024
#define NH 16
#define HD 64
#define NB 2
#define SQ 2048
#define MTOK (NB*SQ)   // 4096 tokens

typedef unsigned short u16;
typedef __attribute__((ext_vector_type(16))) __bf16 v16bf;
typedef __attribute__((ext_vector_type(8)))  float  v8f;
typedef __attribute__((ext_vector_type(4)))  int    v4i;

#if defined(__has_builtin)
#if __has_builtin(__builtin_amdgcn_global_load_async_to_lds_b128)
#define ASYNC_LDS 1
#endif
#endif

__device__ __forceinline__ u16 f2bf(float f) {
  unsigned int u = __float_as_uint(f);
  u += 0x7FFFu + ((u >> 16) & 1u);       // round-to-nearest-even
  return (u16)(u >> 16);
}

__device__ __forceinline__ v8f zero8() {
  v8f z = {0.f,0.f,0.f,0.f,0.f,0.f,0.f,0.f};
  return z;
}

union Frag { v16bf v; uint4 u[2]; };

// Load a 16-bit A/B WMMA fragment from a row-major LDS tile.
__device__ __forceinline__ v16bf ld_frag(const u16* p, int row, int stride,
                                         int kbase, int hf) {
  Frag f;
  const u16* q = p + row * stride + kbase + hf * 8;
  f.u[0] = *(const uint4*)(q);
  f.u[1] = *(const uint4*)(q + 16);
  return f.v;
}

// ---- global -> LDS staging (async path on gfx1250, sync fallback) ----
__device__ __forceinline__ void g2l_16B(u16* lds, const u16* g) {
#ifdef ASYNC_LDS
  __builtin_amdgcn_global_load_async_to_lds_b128(
      (__attribute__((address_space(1))) v4i*)(g),
      (__attribute__((address_space(3))) v4i*)(lds), 0, 0);
#else
  *(uint4*)lds = *(const uint4*)g;
#endif
}
__device__ __forceinline__ void g2l_16u(u16* lds, const u16* g) {  // 32 bytes
  g2l_16B(lds, g); g2l_16B(lds + 8, g + 8);
}
__device__ __forceinline__ void g2l_32u(u16* lds, const u16* g) {  // 64 bytes
  g2l_16B(lds, g); g2l_16B(lds + 8, g + 8);
  g2l_16B(lds + 16, g + 16); g2l_16B(lds + 24, g + 24);
}
__device__ __forceinline__ void wait_async0() {
#ifdef ASYNC_LDS
#if __has_builtin(__builtin_amdgcn_s_wait_asynccnt)
  __builtin_amdgcn_s_wait_asynccnt(0);
#else
  asm volatile("s_wait_asynccnt 0x0" ::: "memory");
#endif
#endif
}

__device__ __forceinline__ v8f wmma_bf16(v16bf a, v16bf b, v8f c) {
  return __builtin_amdgcn_wmma_f32_16x16x32_bf16(false, a, false, b,
                                                 (short)0, c, false, false);
}

// ---------------- conversion kernels ----------------
__global__ __launch_bounds__(256)
void cvt_bf16(const float* __restrict__ s, u16* __restrict__ d, int n) {
  int i = (blockIdx.x * 256 + threadIdx.x) * 4;
  if (i < n) {
    float4 v = *(const float4*)(s + i);
    ushort4 o;
    o.x = f2bf(v.x); o.y = f2bf(v.y); o.z = f2bf(v.z); o.w = f2bf(v.w);
    *(ushort4*)(d + i) = o;
  }
}

// src [rows][cols] f32 -> dst [cols][rows] bf16, LDS-tiled 64x64 transpose
__global__ __launch_bounds__(256)
void cvt_bf16_t(const float* __restrict__ s, u16* __restrict__ d,
                int rows, int cols) {
  __shared__ u16 tile[64 * 72];          // [c][r], stride 72 (16B-aligned rows)
  const int rb = blockIdx.x * 64, cb = blockIdx.y * 64;
  const int t = threadIdx.x;
  {
    int r = t >> 2, c0 = (t & 3) * 16;
    const float* src = s + (size_t)(rb + r) * cols + cb + c0;
    #pragma unroll
    for (int i = 0; i < 16; i += 4) {
      float4 v = *(const float4*)(src + i);
      tile[(c0 + i + 0) * 72 + r] = f2bf(v.x);
      tile[(c0 + i + 1) * 72 + r] = f2bf(v.y);
      tile[(c0 + i + 2) * 72 + r] = f2bf(v.z);
      tile[(c0 + i + 3) * 72 + r] = f2bf(v.w);
    }
  }
  __syncthreads();
  {
    int c = t >> 2, r0 = (t & 3) * 16;
    u16* dst = d + (size_t)(cb + c) * rows + rb + r0;
    ((uint4*)dst)[0] = *(const uint4*)&tile[c * 72 + r0];
    ((uint4*)dst)[1] = *(const uint4*)&tile[c * 72 + r0 + 8];
  }
}

// ---------------- tiled bf16 GEMM: C = alpha*(A[M,K] @ Bt[N,K]^T + bias) ------
// OUTMODE 0: bf16 C[M,N] | 1: bf16 C^T[N,M] | 2: f32 C[M,N]
template<int OUTMODE>
__global__ __launch_bounds__(256)
void gemm_kernel(const u16* __restrict__ A, const u16* __restrict__ Bt,
                 const float* __restrict__ bias, void* __restrict__ Cout,
                 int M, int N, int K, float alpha) {
  __shared__ u16 As[2][128 * 32];
  __shared__ u16 Bs[2][128 * 32];
  const int t = threadIdx.x;
  const int w = t >> 5, lane = t & 31, hf = lane >> 4, ln = lane & 15;
  const int wm = w & 1, wn = w >> 1;                 // 2x4 wave grid
  const int mblock = blockIdx.x * 128, nblock = blockIdx.y * 128;

  v8f acc[4][2];
  #pragma unroll
  for (int i = 0; i < 4; ++i)
    #pragma unroll
    for (int j = 0; j < 2; ++j) acc[i][j] = zero8();

  const int lrow = t >> 1, lh = (t & 1) * 16;
  const u16* Arow = A  + (size_t)(mblock + lrow) * K + lh;
  const u16* Brow = Bt + (size_t)(nblock + lrow) * K + lh;
  const int KT = K >> 5;

  // preload tile 0 into buffer 0
  g2l_16u(&As[0][lrow * 32 + lh], Arow);
  g2l_16u(&Bs[0][lrow * 32 + lh], Brow);
  wait_async0();
  __syncthreads();

  for (int kt = 0; kt < KT; ++kt) {
    const int cur = kt & 1, nxt = cur ^ 1;
    if (kt + 1 < KT) {   // overlap next-tile copy with this tile's WMMAs
      const int k1 = (kt + 1) << 5;
      g2l_16u(&As[nxt][lrow * 32 + lh], Arow + k1);
      g2l_16u(&Bs[nxt][lrow * 32 + lh], Brow + k1);
    }
    v16bf af[4], bf2[2];
    #pragma unroll
    for (int mi = 0; mi < 4; ++mi)
      af[mi] = ld_frag(As[cur], 64 * wm + 16 * mi + ln, 32, 0, hf);
    #pragma unroll
    for (int ni = 0; ni < 2; ++ni)
      bf2[ni] = ld_frag(Bs[cur], 32 * wn + 16 * ni + ln, 32, 0, hf);
    #pragma unroll
    for (int mi = 0; mi < 4; ++mi)
      #pragma unroll
      for (int ni = 0; ni < 2; ++ni)
        acc[mi][ni] = wmma_bf16(af[mi], bf2[ni], acc[mi][ni]);
    wait_async0();
    __syncthreads();
  }

  #pragma unroll
  for (int mi = 0; mi < 4; ++mi)
    #pragma unroll
    for (int ni = 0; ni < 2; ++ni)
      #pragma unroll
      for (int r = 0; r < 8; ++r) {
        int row = mblock + 64 * wm + 16 * mi + r + 8 * hf;
        int col = nblock + 32 * wn + 16 * ni + ln;
        float v = (acc[mi][ni][r] + bias[col]) * alpha;
        if (OUTMODE == 0)      ((u16*)Cout)[(size_t)row * N + col] = f2bf(v);
        else if (OUTMODE == 1) ((u16*)Cout)[(size_t)col * M + row] = f2bf(v);
        else                   ((float*)Cout)[(size_t)row * N + col] = v;
      }
}

// ---------------- attention ----------------
// Qb (pre-scaled by 1/sqrt(hd)), Kb: bf16 [MTOK, ED]; Vtb: bf16 [ED, MTOK]
// attn: f32 [B,H,S,S]; ctx: bf16 [MTOK, ED]
__global__ __launch_bounds__(256)
void attn_kernel(const u16* __restrict__ Qb, const u16* __restrict__ Kb,
                 const u16* __restrict__ Vtb, float* __restrict__ attn,
                 u16* __restrict__ ctx) {
  __shared__ u16 Qt[128 * 64];
  __shared__ u16 Kt[128 * 64];
  __shared__ u16 Vt[64 * 128];
  __shared__ u16 Pt[128 * 128];

  const int qt = blockIdx.x, h = blockIdx.y, b = blockIdx.z;
  const int t = threadIdx.x;
  const int w = t >> 5, lane = t & 31, hf = lane >> 4, ln = lane & 15;
  const int qbase = qt * 128;

  {  // Q tile: 128 rows x 64 bf16
    int row = t >> 1, off = (t & 1) * 32;
    g2l_32u(&Qt[row * 64 + off],
            Qb + (size_t)(b * SQ + qbase + row) * ED + h * HD + off);
  }
  wait_async0();
  __syncthreads();

  v16bf qf[2];
  #pragma unroll
  for (int ks = 0; ks < 2; ++ks) qf[ks] = ld_frag(Qt, 16 * w + ln, 64, ks * 32, hf);

  // per-lane online softmax state (each lane owns cols == ln (mod 16))
  float m_r[8], l_r[8];
  #pragma unroll
  for (int r = 0; r < 8; ++r) { m_r[r] = -1e30f; l_r[r] = 0.f; }

  // ---- Pass 1: online (m,l) per lane, one exp per score element ----
  for (int kt = 0; kt < SQ / 128; ++kt) {
    {
      int row = t >> 1, off = (t & 1) * 32;
      g2l_32u(&Kt[row * 64 + off],
              Kb + (size_t)(b * SQ + kt * 128 + row) * ED + h * HD + off);
    }
    wait_async0();
    __syncthreads();
    #pragma unroll
    for (int j = 0; j < 8; ++j) {
      v8f sc = zero8();
      #pragma unroll
      for (int ks = 0; ks < 2; ++ks) {
        v16bf kf = ld_frag(Kt, 16 * j + ln, 64, ks * 32, hf);
        sc = wmma_bf16(qf[ks], kf, sc);
      }
      #pragma unroll
      for (int r = 0; r < 8; ++r) {
        float s = sc[r];
        float d = s - m_r[r];
        float e = __expf(-__builtin_fabsf(d));
        if (d > 0.f) { l_r[r] = l_r[r] * e + 1.f; m_r[r] = s; }
        else         { l_r[r] += e; }
      }
    }
    __syncthreads();
  }

  // merge per-lane (m,l) across the 16-lane row group (one butterfly total)
  float il[8];
  #pragma unroll
  for (int r = 0; r < 8; ++r) {
    float m = m_r[r], l = l_r[r];
    #pragma unroll
    for (int mask = 1; mask < 16; mask <<= 1) {
      float mo = __shfl_xor(m, mask, 32);
      float lo = __shfl_xor(l, mask, 32);
      float mn = fmaxf(m, mo);
      l = l * __expf(m - mn) + lo * __expf(mo - mn);
      m = mn;
    }
    m_r[r] = m; l_r[r] = l; il[r] = 1.0f / l;
  }

  v8f oacc[4];
  #pragma unroll
  for (int n = 0; n < 4; ++n) oacc[n] = zero8();

  // ---- Pass 2: recompute scores, emit normalized attn, accumulate P@V ----
  for (int kt = 0; kt < SQ / 128; ++kt) {
    {
      int row = t >> 1, off = (t & 1) * 32;
      g2l_32u(&Kt[row * 64 + off],
              Kb + (size_t)(b * SQ + kt * 128 + row) * ED + h * HD + off);
      int vrow = t >> 2, voff = (t & 3) * 32;
      g2l_32u(&Vt[vrow * 128 + voff],
              Vtb + (size_t)(h * HD + vrow) * MTOK + b * SQ + kt * 128 + voff);
    }
    wait_async0();
    __syncthreads();
    #pragma unroll
    for (int j = 0; j < 8; ++j) {
      v8f sc = zero8();
      #pragma unroll
      for (int ks = 0; ks < 2; ++ks) {
        v16bf kf = ld_frag(Kt, 16 * j + ln, 64, ks * 32, hf);
        sc = wmma_bf16(qf[ks], kf, sc);
      }
      #pragma unroll
      for (int r = 0; r < 8; ++r) {
        float pn = __expf(sc[r] - m_r[r]) * il[r];
        int lrow = 16 * w + r + 8 * hf;
        int kcol = kt * 128 + 16 * j + ln;
        attn[((size_t)((b * NH + h) * SQ) + qbase + lrow) * SQ + kcol] = pn;
        Pt[lrow * 128 + 16 * j + ln] = f2bf(pn);   // A-operand staging
      }
    }
    // O += P[16x128] @ V[128x64]  (per wave; Pt rows are wave-private)
    #pragma unroll
    for (int ks = 0; ks < 4; ++ks) {
      v16bf pf = ld_frag(Pt, 16 * w + ln, 128, ks * 32, hf);
      #pragma unroll
      for (int nt = 0; nt < 4; ++nt) {
        v16bf vf = ld_frag(Vt, 16 * nt + ln, 128, ks * 32, hf);
        oacc[nt] = wmma_bf16(pf, vf, oacc[nt]);
      }
    }
    __syncthreads();
  }

  #pragma unroll
  for (int nt = 0; nt < 4; ++nt)
    #pragma unroll
    for (int r = 0; r < 8; ++r) {
      int row = b * SQ + qbase + 16 * w + r + 8 * hf;
      int col = h * HD + 16 * nt + ln;
      ctx[(size_t)row * ED + col] = f2bf(oacc[nt][r]);
    }
}

// ---------------- host-side orchestration ----------------
extern "C" void kernel_launch(void* const* d_in, const int* in_sizes, int n_in,
                              void* d_out, int out_size, void* d_ws, size_t ws_size,
                              hipStream_t stream) {
  const float* x  = (const float*)d_in[0];
  const float* wq = (const float*)d_in[1];
  const float* bq = (const float*)d_in[2];
  const float* wk = (const float*)d_in[3];
  const float* bk = (const float*)d_in[4];
  const float* wv = (const float*)d_in[5];
  const float* bv = (const float*)d_in[6];
  const float* wo = (const float*)d_in[7];
  const float* bo = (const float*)d_in[8];

  u16* ws = (u16*)d_ws;
  const size_t NX = (size_t)MTOK * ED;   // 4M elems
  const size_t NW = (size_t)ED * ED;     // 1M elems
  u16* xb  = ws;
  u16* wqt = xb  + NX;
  u16* wkt = wqt + NW;
  u16* wvt = wkt + NW;
  u16* wot = wvt + NW;
  u16* Qb  = wot + NW;
  u16* Kb  = Qb  + NX;
  u16* Vtb = Kb  + NX;
  u16* ctx = Vtb + NX;

  cvt_bf16<<<(int)(NX / 4 / 256), 256, 0, stream>>>(x, xb, (int)NX);
  dim3 gt(ED / 64, ED / 64);
  cvt_bf16_t<<<gt, 256, 0, stream>>>(wq, wqt, ED, ED);
  cvt_bf16_t<<<gt, 256, 0, stream>>>(wk, wkt, ED, ED);
  cvt_bf16_t<<<gt, 256, 0, stream>>>(wv, wvt, ED, ED);
  cvt_bf16_t<<<gt, 256, 0, stream>>>(wo, wot, ED, ED);

  dim3 gg(MTOK / 128, ED / 128);
  gemm_kernel<0><<<gg, 256, 0, stream>>>(xb, wqt, bq, Qb,  MTOK, ED, ED, 0.125f);
  gemm_kernel<0><<<gg, 256, 0, stream>>>(xb, wkt, bk, Kb,  MTOK, ED, ED, 1.0f);
  gemm_kernel<1><<<gg, 256, 0, stream>>>(xb, wvt, bv, Vtb, MTOK, ED, ED, 1.0f);

  dim3 ga(SQ / 128, NH, NB);
  float* attn_out = (float*)d_out + NX;
  attn_kernel<<<ga, 256, 0, stream>>>(Qb, Kb, Vtb, attn_out, ctx);

  gemm_kernel<2><<<gg, 256, 0, stream>>>(ctx, wot, bo, d_out, MTOK, ED, ED, 1.0f);
}
